// _fasterRCNN_FCN_26929444946236
// MI455X (gfx1250) — compile-verified, matrix operand order checked by reading, not run on hardware
//
#include <hip/hip_runtime.h>
#include <hip/hip_bf16.h>
#include <math.h>

// ---------------------------------------------------------------------------
// Faster-RCNN-FCN head for MI455X (gfx1250), wave32 + WMMA.
//
// Stages:
//   0) zero fcn_loss accumulator
//   1) convert W_cls|W_bbox -> combined N-major f16 matrix WT[80][25088]
//   2) roi_pool(PS=7, scale=1/16) -> pooled_flat f16 A[96][25088]
//   3) fcn_loss reduction over GT boxes (atomicAdd into d_out[7680])
//   4) GEMM (96x25088)x(25088x80) via v_wmma_f32_16x16x32_f16,
//      K split into 49 chunks of 512 (1470 waves) -> partial tiles in ws
//   4b) deterministic K-partial reduction + bias
//      -> cls_score (ws) and bbox_pred (d_out[1536..7680))
//   5) bbox transform -> predicted boxes (ws)
//   6) fcn_score: per (roi,class) roi_pool(PS2=50) max-sum of sq-err (ws)
//   7) fused softmax(-fcn) -> softmax(cls*p) -> cls_prob (d_out[0..1536))
// ---------------------------------------------------------------------------

typedef __attribute__((ext_vector_type(16))) _Float16 v16h;
typedef __attribute__((ext_vector_type(8)))  _Float16 v8h;
typedef __attribute__((ext_vector_type(8)))  float    v8f;

#define NCLS_ 16
#define PS_   7
#define PS2_  50
#define BB_   2
#define NN_   48
#define CC_   512
#define FH_   16
#define FW_   16
#define IH_   256
#define IW_   256
#define KK_   (CC_ * PS_ * PS_)   /* 25088 */
#define RR_   (BB_ * NN_)         /* 96    */
#define NW_   80                  /* 16 cls + 64 bbox */
#define KGT_  10
#define KSPLIT_ 49                /* 25088 = 49 * 512 */
#define KCHUNK_ 512               /* 16 wmma steps per chunk */

// ---------------------------------------------------------------- init ------
__global__ void k_init(float* loss) {
  if (threadIdx.x == 0 && blockIdx.x == 0) *loss = 0.0f;
}

// ---------------------------------------------- weight convert + transpose --
// WT[n][k] (f16, N-major) so GEMM B-fragment loads are contiguous b128s.
__global__ void k_convw(const float* __restrict__ Wc,
                        const float* __restrict__ Wb,
                        _Float16* __restrict__ WT) {
  long idx = (long)blockIdx.x * blockDim.x + threadIdx.x;
  const long total = (long)NW_ * KK_;
  if (idx >= total) return;
  int n = (int)(idx / KK_);
  int k = (int)(idx % KK_);
  float v = (n < NCLS_) ? Wc[(long)k * NCLS_ + n]
                        : Wb[(long)k * 64 + (n - NCLS_)];
  WT[(long)n * KK_ + k] = (_Float16)v;
}

// --------------------------------------------------------- roi_pool PS=7 ---
__global__ void k_roipool7(const float* __restrict__ feat,
                           const float* __restrict__ rois,
                           _Float16* __restrict__ A) {
  long idx = (long)blockIdx.x * blockDim.x + threadIdx.x;
  const long total = (long)RR_ * CC_ * PS_ * PS_;
  if (idx >= total) return;
  int r   = (int)(idx / (CC_ * PS_ * PS_));
  int rem = (int)(idx % (CC_ * PS_ * PS_));
  int c   = rem / (PS_ * PS_);
  int s   = rem % (PS_ * PS_);
  int b   = r / NN_;

  const float* bx = rois + (long)r * 5;
  int x1 = (int)roundf(bx[1] * (1.0f / 16.0f));
  int y1 = (int)roundf(bx[2] * (1.0f / 16.0f));
  int x2 = (int)roundf(bx[3] * (1.0f / 16.0f));
  int y2 = (int)roundf(bx[4] * (1.0f / 16.0f));
  int lenx = max(x2 - x1 + 1, 1);
  int leny = max(y2 - y1 + 1, 1);
  float bsx = (float)lenx / PS_;
  float bsy = (float)leny / PS_;
  int px = s % PS_, py = s / PS_;
  int xlo = max((int)floorf(px * bsx), 0);
  int xhi = min((int)ceilf((px + 1) * bsx), lenx);
  int ylo = max((int)floorf(py * bsy), 0);
  int yhi = min((int)ceilf((py + 1) * bsy), leny);

  const float* fp = feat + ((long)b * CC_ + c) * FH_ * FW_;
  float m = -__builtin_inff();
  for (int yy = ylo; yy < yhi; ++yy) {
    int gy = y1 + yy;
    if (gy < 0 || gy >= FH_) continue;
    for (int xx = xlo; xx < xhi; ++xx) {
      int gx = x1 + xx;
      if (gx < 0 || gx >= FW_) continue;
      m = fmaxf(m, fp[gy * FW_ + gx]);
    }
  }
  if (m == -__builtin_inff()) m = 0.0f;
  A[(long)r * KK_ + c * (PS_ * PS_) + s] = (_Float16)m;
}

// --------------------------------------------------------------- fcn_loss ---
__global__ void k_fcnloss(const float* __restrict__ im,
                          const float* __restrict__ yf,
                          const float* __restrict__ gt,
                          const int* __restrict__ nb,
                          float* __restrict__ loss) {
  int blk = blockIdx.x;               // B*KGT blocks
  int b = blk / KGT_, k = blk % KGT_;
  const float* g = gt + ((long)b * KGT_ + k) * 5;
  int gx1 = (int)g[0], gy1 = (int)g[1];
  int gx2 = (int)g[2], gy2 = (int)g[3];
  int cls = (int)g[4];
  int xlo = max(gx1, 0), xhi = min(gx2, IW_);
  int ylo = max(gy1, 0), yhi = min(gy2, IH_);
  int w = max(xhi - xlo, 0), h = max(yhi - ylo, 0);
  int npix = w * h;

  int tid = threadIdx.x;
  float sum = 0.0f;
  for (int p = tid; p < npix; p += blockDim.x) {
    int y = ylo + p / w;
    int x = xlo + p % w;
    for (int ch = 0; ch < 3; ++ch) {
      float a = im[(((long)b * 3 + ch) * IH_ + y) * IW_ + x];
      float q = yf[(((long)b * (3 * NCLS_) + cls * 3 + ch) * IH_ + y) * IW_ + x];
      float d = a - q;
      sum += d * d;
    }
  }
  __shared__ float sd[256];
  sd[tid] = sum;
  __syncthreads();
  for (int st = 128; st > 0; st >>= 1) {
    if (tid < st) sd[tid] += sd[tid + st];
    __syncthreads();
  }
  if (tid == 0) {
    if (k < nb[b]) {
      float denom = 3.0f * (float)max((gy2 - gy1) * (gx2 - gx1), 1);
      float tot = (float)(nb[0] + nb[1]);
      atomicAdd(loss, sd[0] / denom / tot / 10000.0f);
    }
  }
}

// ------------------------------------------------------------ WMMA GEMM ----
// part[kz][96][80] = A[:, kz*512:(kz+1)*512] x WT^T chunk.
// One wave per (16x16 tile, K-chunk): grid (6,5,49) = 1470 waves, so the
// memory-bound GEMM can actually stretch across the WGPs instead of leaving
// the HBM/L2 pipes idle (30 waves can't feed 23.3 TB/s).
// A-frag & B-frag per-lane layout (ISA 16-bit 16x32): lane half hf selects
// K offset +8*hf; the 16 values are two contiguous 8-element runs at
// K=hf*8 and K=16+hf*8 -> two b128 loads each.
__global__ void k_gemm_wmma(const _Float16* __restrict__ A,
                            const _Float16* __restrict__ BT,
                            float* __restrict__ part) {   // [49][96][80]
  const int mt   = blockIdx.x;      // 0..5
  const int nt   = blockIdx.y;      // 0..4
  const int kz   = blockIdx.z;      // 0..48
  const int lane = threadIdx.x;     // 0..31 (wave32)
  const int col  = lane & 15;
  const int hf   = lane >> 4;

  const _Float16* Ap = A  + (long)(mt * 16 + col) * KK_ + kz * KCHUNK_ + hf * 8;
  const _Float16* Bp = BT + (long)(nt * 16 + col) * KK_ + kz * KCHUNK_ + hf * 8;

  v8f acc = {};
  for (int k0 = 0; k0 < KCHUNK_; k0 += 32) {
    v8h a0 = *(const v8h*)(Ap + k0);
    v8h a1 = *(const v8h*)(Ap + k0 + 16);
    v8h b0 = *(const v8h*)(Bp + k0);
    v8h b1 = *(const v8h*)(Bp + k0 + 16);
    if (k0 + 32 < KCHUNK_) {
      __builtin_prefetch(Ap + k0 + 64, 0, 1);   // global_prefetch path
      __builtin_prefetch(Bp + k0 + 64, 0, 1);
    }
    v16h av = __builtin_shufflevector(a0, a1, 0, 1, 2, 3, 4, 5, 6, 7,
                                      8, 9, 10, 11, 12, 13, 14, 15);
    v16h bv = __builtin_shufflevector(b0, b1, 0, 1, 2, 3, 4, 5, 6, 7,
                                      8, 9, 10, 11, 12, 13, 14, 15);
    acc = __builtin_amdgcn_wmma_f32_16x16x32_f16(
        /*neg_a=*/false, av, /*neg_b=*/false, bv,
        /*c_mod=*/(short)0, acc, /*reuse_a=*/false, /*reuse_b=*/false);
  }

  const int n_g = nt * 16 + col;
  for (int j = 0; j < 8; ++j) {
    int m_g = mt * 16 + j + 8 * hf;           // C/D layout: VGPR j -> M=j+8*hf
    part[((long)kz * RR_ + m_g) * NW_ + n_g] = acc[j];
  }
}

// ------------------------------------- deterministic K-partial reduction ---
__global__ void k_gemm_reduce(const float* __restrict__ part,
                              const float* __restrict__ bcls,
                              const float* __restrict__ bbbox,
                              float* __restrict__ scores,     // ws: 96x16
                              float* __restrict__ out_bbox) { // d_out+1536
  int idx = blockIdx.x * blockDim.x + threadIdx.x;
  if (idx >= RR_ * NW_) return;
  int m = idx / NW_, n = idx % NW_;
  float s = 0.0f;
  for (int kz = 0; kz < KSPLIT_; ++kz)        // fixed order => deterministic
    s += part[((long)kz * RR_ + m) * NW_ + n];
  s += (n < NCLS_) ? bcls[n] : bbbox[n - NCLS_];
  if (n < NCLS_) scores[m * NCLS_ + n] = s;
  else           out_bbox[m * 64 + (n - NCLS_)] = s;
}

// --------------------------------------------------------- bbox transform --
__global__ void k_bbox(const float* __restrict__ bbox_pred, // d_out+1536
                       const float* __restrict__ rois,
                       const float* __restrict__ im_info,
                       float* __restrict__ pred) {          // ws: [r*16+c][4]
  int idx = blockIdx.x * blockDim.x + threadIdx.x;
  if (idx >= RR_ * NCLS_) return;
  int r = idx / NCLS_, c = idx % NCLS_;
  int b = r / NN_;
  float d0 = bbox_pred[r * 64 + c * 4 + 0] * 0.1f;
  float d1 = bbox_pred[r * 64 + c * 4 + 1] * 0.1f;
  float d2 = bbox_pred[r * 64 + c * 4 + 2] * 0.2f;
  float d3 = bbox_pred[r * 64 + c * 4 + 3] * 0.2f;
  const float* bx = rois + (long)r * 5;
  float w  = bx[3] - bx[1] + 1.0f;
  float h  = bx[4] - bx[2] + 1.0f;
  float cx = bx[1] + 0.5f * w;
  float cy = bx[2] + 0.5f * h;
  float pcx = d0 * w + cx, pcy = d1 * h + cy;
  float pw = expf(d2) * w, ph = expf(d3) * h;
  float H = im_info[b * 3 + 0], W = im_info[b * 3 + 1];
  float sc = im_info[2];                     // im_info[0,2]
  float x1 = fminf(fmaxf(pcx - 0.5f * pw, 0.0f), W - 1.0f) / sc;
  float y1 = fminf(fmaxf(pcy - 0.5f * ph, 0.0f), H - 1.0f) / sc;
  float x2 = fminf(fmaxf(pcx + 0.5f * pw, 0.0f), W - 1.0f) / sc;
  float y2 = fminf(fmaxf(pcy + 0.5f * ph, 0.0f), H - 1.0f) / sc;
  pred[(long)idx * 4 + 0] = x1;
  pred[(long)idx * 4 + 1] = y1;
  pred[(long)idx * 4 + 2] = x2;
  pred[(long)idx * 4 + 3] = y2;
}

// ------------------------------------------------------------- fcn_score ---
// One block per (roi, class); roi_pool(PS2=50) max per bin of squared error,
// summed over 2500 bins x 3 channels.
__global__ void k_fcnscore(const float* __restrict__ im,
                           const float* __restrict__ yf,
                           const float* __restrict__ pred,
                           float* __restrict__ fs) {        // ws: [r][16]
  int blk = blockIdx.x;                 // RR_*NCLS_ blocks
  int r = blk / NCLS_, c = blk % NCLS_, b = r / NN_;
  const float* p = pred + (long)blk * 4;
  int x1 = (int)roundf(p[0]), y1 = (int)roundf(p[1]);
  int x2 = (int)roundf(p[2]), y2 = (int)roundf(p[3]);
  int lenx = max(x2 - x1 + 1, 1), leny = max(y2 - y1 + 1, 1);
  float bsx = (float)lenx / PS2_, bsy = (float)leny / PS2_;

  int tid = threadIdx.x;
  float sum = 0.0f;
  const int ntask = 3 * PS2_ * PS2_;
  for (int t = tid; t < ntask; t += blockDim.x) {
    int ch = t / (PS2_ * PS2_);
    int s  = t % (PS2_ * PS2_);
    int py = s / PS2_, px = s % PS2_;
    int xlo = max((int)floorf(px * bsx), 0);
    int xhi = min((int)ceilf((px + 1) * bsx), lenx);
    int ylo = max((int)floorf(py * bsy), 0);
    int yhi = min((int)ceilf((py + 1) * bsy), leny);
    const float* ip = im + ((long)b * 3 + ch) * IH_ * IW_;
    const float* yp = yf + ((long)b * (3 * NCLS_) + c * 3 + ch) * IH_ * IW_;
    float m = -__builtin_inff();
    for (int yy = ylo; yy < yhi; ++yy) {
      int gy = y1 + yy;
      if (gy < 0 || gy >= IH_) continue;
      for (int xx = xlo; xx < xhi; ++xx) {
        int gx = x1 + xx;
        if (gx < 0 || gx >= IW_) continue;
        float d = ip[gy * IW_ + gx] - yp[gy * IW_ + gx];
        m = fmaxf(m, d * d);
      }
    }
    if (m != -__builtin_inff()) sum += m;
  }
  __shared__ float sd[256];
  sd[tid] = sum;
  __syncthreads();
  for (int st = 128; st > 0; st >>= 1) {
    if (tid < st) sd[tid] += sd[tid + st];
    __syncthreads();
  }
  if (tid == 0) fs[blk] = sd[0];
}

// ---------------------------------------------------------- fused softmax --
__global__ void k_softmax(const float* __restrict__ scores,
                          const float* __restrict__ fs,
                          float* __restrict__ out) {        // d_out[0..1536)
  int r = blockIdx.x * blockDim.x + threadIdx.x;
  if (r >= RR_) return;
  float f[NCLS_], z[NCLS_];
  float mx = -__builtin_inff();
  for (int c = 0; c < NCLS_; ++c) {
    f[c] = -fs[r * NCLS_ + c];
    mx = fmaxf(mx, f[c]);
  }
  float s1 = 0.0f;
  for (int c = 0; c < NCLS_; ++c) { f[c] = expf(f[c] - mx); s1 += f[c]; }
  float mz = -__builtin_inff();
  for (int c = 0; c < NCLS_; ++c) {
    z[c] = scores[r * NCLS_ + c] * (f[c] / s1);
    mz = fmaxf(mz, z[c]);
  }
  float s2 = 0.0f;
  for (int c = 0; c < NCLS_; ++c) { z[c] = expf(z[c] - mz); s2 += z[c]; }
  for (int c = 0; c < NCLS_; ++c) out[r * NCLS_ + c] = z[c] / s2;
}

// ---------------------------------------------------------------------------
extern "C" void kernel_launch(void* const* d_in, const int* in_sizes, int n_in,
                              void* d_out, int out_size, void* d_ws, size_t ws_size,
                              hipStream_t stream) {
  const float* im_data   = (const float*)d_in[0];
  const float* y_fcn     = (const float*)d_in[1];
  const float* base_feat = (const float*)d_in[2];
  const float* rois      = (const float*)d_in[3];
  const float* im_info   = (const float*)d_in[4];
  const float* gt_boxes  = (const float*)d_in[5];
  const float* W_cls     = (const float*)d_in[6];
  const float* b_cls     = (const float*)d_in[7];
  const float* W_bbox    = (const float*)d_in[8];
  const float* b_bbox    = (const float*)d_in[9];
  const int*   num_boxes = (const int*)d_in[10];

  float* out = (float*)d_out;
  float* out_bbox = out + (long)RR_ * NCLS_;       // [1536..7680)
  float* out_loss = out + (long)RR_ * NCLS_ + (long)RR_ * 64;  // [7680]

  // workspace carve-up
  char* ws = (char*)d_ws;
  _Float16* Apool = (_Float16*)ws;                               // 96*25088 f16
  size_t off = (size_t)RR_ * KK_ * sizeof(_Float16);
  _Float16* WT = (_Float16*)(ws + off);                          // 80*25088 f16
  off += (size_t)NW_ * KK_ * sizeof(_Float16);
  float* scores = (float*)(ws + off);                            // 96*16 f32
  off += (size_t)RR_ * NCLS_ * sizeof(float);
  float* fscore = (float*)(ws + off);                            // 96*16 f32
  off += (size_t)RR_ * NCLS_ * sizeof(float);
  float* pred = (float*)(ws + off);                              // 96*16*4 f32
  off += (size_t)RR_ * NCLS_ * 4 * sizeof(float);
  float* part = (float*)(ws + off);                              // 49*96*80 f32

  k_init<<<1, 1, 0, stream>>>(out_loss);

  {
    long total = (long)NW_ * KK_;
    k_convw<<<(int)((total + 255) / 256), 256, 0, stream>>>(W_cls, W_bbox, WT);
  }
  {
    long total = (long)RR_ * CC_ * PS_ * PS_;
    k_roipool7<<<(int)((total + 255) / 256), 256, 0, stream>>>(base_feat, rois, Apool);
  }
  k_fcnloss<<<BB_ * KGT_, 256, 0, stream>>>(im_data, y_fcn, gt_boxes, num_boxes, out_loss);

  k_gemm_wmma<<<dim3(RR_ / 16, NW_ / 16, KSPLIT_), 32, 0, stream>>>(Apool, WT, part);

  k_gemm_reduce<<<(RR_ * NW_ + 255) / 256, 256, 0, stream>>>(
      part, b_cls, b_bbox, scores, out_bbox);

  k_bbox<<<(RR_ * NCLS_ + 127) / 128, 128, 0, stream>>>(out_bbox, rois, im_info, pred);

  k_fcnscore<<<RR_ * NCLS_, 256, 0, stream>>>(im_data, y_fcn, pred, fscore);

  k_softmax<<<1, 128, 0, stream>>>(scores, fscore, out);
}
